// MultiHeadAttention_62010737819867
// MI455X (gfx1250) — matrix operand order repaired and checked by reading, not compiled, not run on hardware
//
#include <hip/hip_runtime.h>

#define DEV_INLINE __device__ __forceinline__

typedef __bf16 bf16_t;
typedef __attribute__((ext_vector_type(16))) __bf16 v16bf;
typedef __attribute__((ext_vector_type(8)))  __bf16 v8bf;
typedef __attribute__((ext_vector_type(8)))  float  v8f;
typedef unsigned int u32x4 __attribute__((ext_vector_type(4)));
typedef int i32x4 __attribute__((ext_vector_type(4)));
typedef int i32x8 __attribute__((ext_vector_type(8)));

#if defined(__has_builtin)
#if __has_builtin(__builtin_amdgcn_tensor_load_to_lds)
#define HAVE_TDM 1
#else
#define HAVE_TDM 0
#endif
#else
#define HAVE_TDM 0
#endif

#if HAVE_TDM && __has_include(<hip/amd_detail/amd_gfx1250_TDM.h>)
#define TDM_ARITY6 1
#else
#define TDM_ARITY6 0
#endif

static constexpr int Bb = 4, Ss = 2048, Dd = 1024, Hh = 16, DEPTH = 64;

DEV_INLINE v16bf cat8(v8bf lo, v8bf hi) {
  return __builtin_shufflevector(lo, hi, 0, 1, 2, 3, 4, 5, 6, 7, 8, 9, 10, 11, 12, 13, 14, 15);
}

// A-fragment: 16x32 bf16 tile, source row-major [16][stride].
// lanes 0-15 -> M=lane, K {0..7, 16..23}; lanes 16-31 -> M=lane-16, K {8..15, 24..31}.
DEV_INLINE v16bf load_frag_a(const bf16_t* base, int stride) {
  const int lane = threadIdx.x & 31;
  const bf16_t* p = base + (lane & 15) * stride + ((lane >> 4) << 3);
  v8bf lo = *(const v8bf*)p;
  v8bf hi = *(const v8bf*)(p + 16);
  return cat8(lo, hi);
}

// B-fragment from B-transposed source (row-major [N][stride] along K).
// lanes 0-15 -> N=lane, K {0..7, 16..23}; lanes 16-31 -> N=lane-16, K {8..15, 24..31}.
DEV_INLINE v16bf load_frag_bt(const bf16_t* base, int stride) {
  const int lane = threadIdx.x & 31;
  const bf16_t* p = base + (lane & 15) * stride + ((lane >> 4) << 3);
  v8bf lo = *(const v8bf*)p;
  v8bf hi = *(const v8bf*)(p + 16);
  return cat8(lo, hi);
}

DEV_INLINE v8f wmma_bf16(v16bf a, v16bf b, v8f c) {
  return __builtin_amdgcn_wmma_f32_16x16x32_bf16(false, a, false, b, (short)0, c, false, false);
}

#if HAVE_TDM
// Issue a TDM 2-D tile load Global->LDS (bf16 elements, row-major tile).
// tile_x elems per row, tile_y rows, row_stride elems; LDS gets 16B pad after
// every 2^(pad_int_code+1) DWORDs (pad_amt_code+1 DWORDs of padding).
DEV_INLINE void tdm_load_2d(unsigned lds_byte_off, const void* gaddr, unsigned tile_x,
                            unsigned tile_y, unsigned row_stride, unsigned pad_int_code,
                            unsigned pad_amt_code) {
  const unsigned long long ga = (unsigned long long)gaddr;
  u32x4 g0;
  g0[0] = 1u;  // count=1 valid descriptor, user mode, no gather
  g0[1] = lds_byte_off;
  g0[2] = (unsigned)(ga & 0xffffffffu);
  g0[3] = (unsigned)((ga >> 32) & 0x01ffffffu) | 0x80000000u;  // type=2 ("image")
  i32x8 g1;
  g1[0] = (int)((1u << 16) |              // data_size = 2 bytes
                (1u << 20) |              // pad_enable
                (pad_int_code << 22) | (pad_amt_code << 25));
  g1[1] = (int)(tile_x << 16);            // tensor_dim0 = tile_x
  g1[2] = (int)(tile_y << 16);            // tensor_dim1 = tile_y
  g1[3] = (int)(tile_x << 16);            // tile_dim0
  g1[4] = (int)tile_y;                    // tile_dim1 (tile_dim2 = 0)
  g1[5] = (int)row_stride;                // tensor_dim0_stride
  g1[6] = 0;
  g1[7] = 0;
  i32x4 g2 = {1, 1, 0, 0};                // tensor_dim2 = 1, tensor_dim3 = 1 (benign)
  i32x4 g3 = {0, 1 << 16, 0, 0};          // tensor_dim4 = 1
#if TDM_ARITY6
  i32x8 gz8 = {0, 0, 0, 0, 0, 0, 0, 0};
  __builtin_amdgcn_tensor_load_to_lds(g0, g1, g2, g3, gz8, 0);
#else
  __builtin_amdgcn_tensor_load_to_lds(g0, g1, g2, g3, 0);
#endif
}
#endif

// ---------------------------------------------------------------------------
// f32 -> bf16 elementwise convert (weights)
// ---------------------------------------------------------------------------
__global__ __launch_bounds__(256) void cvt_f32_bf16_kernel(const float* __restrict__ s,
                                                           bf16_t* __restrict__ d, int n4) {
  const int i = blockIdx.x * blockDim.x + threadIdx.x;
  if (i < n4) {
    float4 f = ((const float4*)s)[i];
    d[i * 4 + 0] = (bf16_t)f.x;
    d[i * 4 + 1] = (bf16_t)f.y;
    d[i * 4 + 2] = (bf16_t)f.z;
    d[i * 4 + 3] = (bf16_t)f.w;
  }
}

// ---------------------------------------------------------------------------
// Y[M,N] = A[M,K] * W[N,K]^T + bias ; A is f32 or bf16, Y is bf16 or f32.
// Block tile 128x128, K-step 64, 8 waves in 2(M)x4(N) grid, 4x2 WMMA subtiles.
// W tile (and A tile when already bf16) staged via TDM tensor_load_to_lds with
// hardware LDS padding (16B pad per 128B row -> stride 72 halves).
// ---------------------------------------------------------------------------
template <typename AT, typename OT>
__global__ __launch_bounds__(256) void gemm_xwt_kernel(const AT* __restrict__ A,
                                                       const bf16_t* __restrict__ W,
                                                       const float* __restrict__ bias,
                                                       OT* __restrict__ Y,
                                                       int M, int N, int K) {
  __shared__ bf16_t smem[2 * 128 * 72];
  bf16_t* As = smem;                                   // LDS byte offset 0
  bf16_t* Ws = smem + 128 * 72;                        // LDS byte offset 18432
  constexpr unsigned kWsOff = 128 * 72 * sizeof(bf16_t);

  const int n0 = blockIdx.x * 128;
  const int m0 = blockIdx.y * 128;
  const int tid = threadIdx.x;
  const int lane = tid & 31;
  const int wave = tid >> 5;
  const int wm = (wave >> 2) * 64;  // wave M offset in tile
  const int wn = (wave & 3) * 32;   // wave N offset in tile

  v8f acc[4][2];
#pragma unroll
  for (int i = 0; i < 4; ++i)
#pragma unroll
    for (int j = 0; j < 2; ++j) {
      v8f z = {0.f, 0.f, 0.f, 0.f, 0.f, 0.f, 0.f, 0.f};
      acc[i][j] = z;
    }

  const int sr = tid >> 1;        // 0..127
  const int sc = (tid & 1) << 5;  // 0 / 32

  for (int k0 = 0; k0 < K; k0 += 64) {
#if HAVE_TDM
    if constexpr (sizeof(AT) == 2) {
      if (wave == 0) {
        tdm_load_2d(0u, A + (size_t)m0 * K + k0, 64, 128, (unsigned)K, 4, 3);
        tdm_load_2d(kWsOff, W + (size_t)n0 * K + k0, 64, 128, (unsigned)K, 4, 3);
        __builtin_amdgcn_s_wait_tensorcnt((short)0);
      }
    } else {
      if (wave == 0)  // async DMA of W tile overlaps manual A staging below
        tdm_load_2d(kWsOff, W + (size_t)n0 * K + k0, 64, 128, (unsigned)K, 4, 3);
      const AT* src = A + (size_t)(m0 + sr) * K + k0 + sc;
      bf16_t* dst = As + sr * 72 + sc;
      const float4* s4 = (const float4*)src;
#pragma unroll
      for (int u = 0; u < 4; ++u) {
        float4 fa = s4[2 * u], fb = s4[2 * u + 1];
        v8bf t;
        t[0] = (bf16_t)fa.x; t[1] = (bf16_t)fa.y; t[2] = (bf16_t)fa.z; t[3] = (bf16_t)fa.w;
        t[4] = (bf16_t)fb.x; t[5] = (bf16_t)fb.y; t[6] = (bf16_t)fb.z; t[7] = (bf16_t)fb.w;
        *(v8bf*)(dst + u * 8) = t;
      }
      if (wave == 0) __builtin_amdgcn_s_wait_tensorcnt((short)0);
    }
#else
    {
      const AT* src = A + (size_t)(m0 + sr) * K + k0 + sc;
      bf16_t* dst = As + sr * 72 + sc;
      if constexpr (sizeof(AT) == 4) {
        const float4* s4 = (const float4*)src;
#pragma unroll
        for (int u = 0; u < 4; ++u) {
          float4 fa = s4[2 * u], fb = s4[2 * u + 1];
          v8bf t;
          t[0] = (bf16_t)fa.x; t[1] = (bf16_t)fa.y; t[2] = (bf16_t)fa.z; t[3] = (bf16_t)fa.w;
          t[4] = (bf16_t)fb.x; t[5] = (bf16_t)fb.y; t[6] = (bf16_t)fb.z; t[7] = (bf16_t)fb.w;
          *(v8bf*)(dst + u * 8) = t;
        }
      } else {
        const v8bf* s8 = (const v8bf*)src;
#pragma unroll
        for (int u = 0; u < 4; ++u) *(v8bf*)(dst + u * 8) = s8[u];
      }
      const bf16_t* wsrc = W + (size_t)(n0 + sr) * K + k0 + sc;
      bf16_t* wdst = Ws + sr * 72 + sc;
      const v8bf* w8 = (const v8bf*)wsrc;
#pragma unroll
      for (int u = 0; u < 4; ++u) *(v8bf*)(wdst + u * 8) = w8[u];
    }
#endif
    __syncthreads();

#pragma unroll
    for (int kk = 0; kk < 2; ++kk) {
      v16bf af[4], bfr[2];
#pragma unroll
      for (int i = 0; i < 4; ++i) af[i] = load_frag_a(As + (wm + i * 16) * 72 + kk * 32, 72);
#pragma unroll
      for (int j = 0; j < 2; ++j) bfr[j] = load_frag_bt(Ws + (wn + j * 16) * 72 + kk * 32, 72);
#pragma unroll
      for (int i = 0; i < 4; ++i)
#pragma unroll
        for (int j = 0; j < 2; ++j) acc[i][j] = wmma_bf16(af[i], bfr[j], acc[i][j]);
    }
    __syncthreads();
  }

  const int ln = lane & 15;
  const int hf = lane >> 4;
#pragma unroll
  for (int j = 0; j < 2; ++j) {
    const int gn = n0 + wn + j * 16 + ln;
    const float bv = bias[gn];
#pragma unroll
    for (int i = 0; i < 4; ++i) {
#pragma unroll
      for (int r = 0; r < 8; ++r) {
        const int gm = m0 + wm + i * 16 + r + hf * 8;
        Y[(size_t)gm * N + gn] = (OT)(acc[i][j][r] + bv);
      }
    }
  }
}

// ---------------------------------------------------------------------------
// Causal flash attention: one block = 128 q rows of one (b,h).
// Q fragments register-resident; KV blocks of 64 staged in LDS (K via TDM,
// V^T manual transpose). Online softmax in registers via shfl reductions.
// ---------------------------------------------------------------------------
__global__ __launch_bounds__(256) void flash_attn_kernel(const bf16_t* __restrict__ Qh,
                                                         const bf16_t* __restrict__ Kh,
                                                         const bf16_t* __restrict__ Vh,
                                                         bf16_t* __restrict__ Oc) {
  __shared__ bf16_t smem[64 * 72 + 64 * 72 + 128 * 72];
  bf16_t* Ks = smem;               // [64][72]  LDS byte off 0
  bf16_t* Vt = smem + 64 * 72;     // [64][72]  V^T (depth x n)
  bf16_t* Ps = smem + 2 * 64 * 72; // [128][72] probabilities (wave-private rows)

  const int qb = blockIdx.x;
  const int b = blockIdx.y >> 4;   // / H
  const int h = blockIdx.y & 15;   // % H
  const size_t base = ((size_t)b * Ss) * Dd + (size_t)h * DEPTH;
  const bf16_t* Qp = Qh + base;
  const bf16_t* Kp = Kh + base;
  const bf16_t* Vp = Vh + base;

  const int tid = threadIdx.x;
  const int lane = tid & 31;
  const int wave = tid >> 5;
  const int ln = lane & 15;
  const int hf = lane >> 4;
  const int q0 = qb * 128;
  const int wr = wave * 16;

  v16bf qf[2];
#pragma unroll
  for (int kk = 0; kk < 2; ++kk)
    qf[kk] = load_frag_a(Qp + (size_t)(q0 + wr) * Dd + kk * 32, Dd);

  v8f of[4];
#pragma unroll
  for (int nd = 0; nd < 4; ++nd) {
    v8f z = {0.f, 0.f, 0.f, 0.f, 0.f, 0.f, 0.f, 0.f};
    of[nd] = z;
  }
  float m_i[8], l_i[8];
#pragma unroll
  for (int r = 0; r < 8; ++r) {
    m_i[r] = -3.0e38f;
    l_i[r] = 0.f;
  }
  const float scl = 0.03125f;  // 1/sqrt(d_model) = 1/32
  const int nblk = qb * 2 + 2;

  for (int j = 0; j < nblk; ++j) {
    const int n0 = j * 64;
#if HAVE_TDM
    if (wave == 0)  // async K-tile DMA overlaps the manual V^T staging
      tdm_load_2d(0u, Kp + (size_t)n0 * Dd, 64, 64, (unsigned)Dd, 4, 3);
#endif
    {
      const int r = tid >> 2;        // 0..63
      const int c = (tid & 3) << 4;  // 0,16,32,48
#if !HAVE_TDM
      const bf16_t* ks = Kp + (size_t)(n0 + r) * Dd + c;
      v8bf k0v = *(const v8bf*)ks;
      v8bf k1v = *(const v8bf*)(ks + 8);
      *(v8bf*)(Ks + r * 72 + c) = k0v;
      *(v8bf*)(Ks + r * 72 + c + 8) = k1v;
#endif
      const bf16_t* vs = Vp + (size_t)(n0 + r) * Dd + c;
      v8bf v0v = *(const v8bf*)vs;
      v8bf v1v = *(const v8bf*)(vs + 8);
#pragma unroll
      for (int e = 0; e < 8; ++e) Vt[(c + e) * 72 + r] = v0v[e];
#pragma unroll
      for (int e = 0; e < 8; ++e) Vt[(c + 8 + e) * 72 + r] = v1v[e];
    }
#if HAVE_TDM
    if (wave == 0) __builtin_amdgcn_s_wait_tensorcnt((short)0);
#endif
    __syncthreads();

    // S = Q K^T : wave's 16 rows x 64 cols
    v8f sf[4];
#pragma unroll
    for (int ns = 0; ns < 4; ++ns) {
      v8f z = {0.f, 0.f, 0.f, 0.f, 0.f, 0.f, 0.f, 0.f};
      sf[ns] = z;
    }
#pragma unroll
    for (int ns = 0; ns < 4; ++ns)
#pragma unroll
      for (int kk = 0; kk < 2; ++kk) {
        v16bf kb = load_frag_bt(Ks + ns * 16 * 72 + kk * 32, 72);
        sf[ns] = wmma_bf16(qf[kk], kb, sf[ns]);
      }

    // scale + causal mask + row max
    const bool domask = (n0 + 63 > q0);
    float rm[8];
#pragma unroll
    for (int r = 0; r < 8; ++r) rm[r] = -3.0e38f;
#pragma unroll
    for (int ns = 0; ns < 4; ++ns)
#pragma unroll
      for (int r = 0; r < 8; ++r) {
        float s = sf[ns][r] * scl;
        if (domask) {
          const int gq = q0 + wr + r + hf * 8;
          const int gn = n0 + ns * 16 + ln;
          if (gn > gq) s = -1.0e9f;
        }
        sf[ns][r] = s;
        rm[r] = fmaxf(rm[r], s);
      }
#pragma unroll
    for (int r = 0; r < 8; ++r)
#pragma unroll
      for (int mm = 1; mm < 16; mm <<= 1)
        rm[r] = fmaxf(rm[r], __shfl_xor(rm[r], mm, 32));

    float alpha[8], rs[8];
#pragma unroll
    for (int r = 0; r < 8; ++r) {
      const float mn = fmaxf(m_i[r], rm[r]);
      alpha[r] = __expf(m_i[r] - mn);
      m_i[r] = mn;
      rs[r] = 0.f;
    }
#pragma unroll
    for (int ns = 0; ns < 4; ++ns)
#pragma unroll
      for (int r = 0; r < 8; ++r) {
        const float p = __expf(sf[ns][r] - m_i[r]);
        rs[r] += p;
        Ps[(wr + r + hf * 8) * 72 + ns * 16 + ln] = (bf16_t)p;
      }
#pragma unroll
    for (int r = 0; r < 8; ++r)
#pragma unroll
      for (int mm = 1; mm < 16; mm <<= 1) rs[r] += __shfl_xor(rs[r], mm, 32);
#pragma unroll
    for (int r = 0; r < 8; ++r) l_i[r] = l_i[r] * alpha[r] + rs[r];
#pragma unroll
    for (int nd = 0; nd < 4; ++nd)
#pragma unroll
      for (int r = 0; r < 8; ++r) of[nd][r] *= alpha[r];

    // O += P V
#pragma unroll
    for (int ks = 0; ks < 2; ++ks) {
      v16bf pa = load_frag_a(Ps + wr * 72 + ks * 32, 72);
#pragma unroll
      for (int nd = 0; nd < 4; ++nd) {
        v16bf vb = load_frag_bt(Vt + nd * 16 * 72 + ks * 32, 72);
        of[nd] = wmma_bf16(pa, vb, of[nd]);
      }
    }
    __syncthreads();
  }

#pragma unroll
  for (int nd = 0; nd < 4; ++nd)
#pragma unroll
    for (int r = 0; r < 8; ++r) {
      const float v = of[nd][r] / l_i[r];
      const int gq = q0 + wr + r + hf * 8;
      const int gn = nd * 16 + ln;
      Oc[((size_t)b * Ss + gq) * Dd + h * DEPTH + gn] = (bf16_t)v;
    }
}

// ---------------------------------------------------------------------------
// Host launcher
// ---------------------------------------------------------------------------
extern "C" void kernel_launch(void* const* d_in, const int* in_sizes, int n_in,
                              void* d_out, int out_size, void* d_ws, size_t ws_size,
                              hipStream_t stream) {
  (void)in_sizes; (void)n_in; (void)out_size; (void)ws_size;
  const float* q = (const float*)d_in[0];
  const float* k = (const float*)d_in[1];
  const float* v = (const float*)d_in[2];
  // d_in[3] = mask (causal, computed analytically in-kernel)
  const float* wq_w = (const float*)d_in[4];
  const float* wq_b = (const float*)d_in[5];
  const float* wk_w = (const float*)d_in[6];
  const float* wk_b = (const float*)d_in[7];
  const float* wv_w = (const float*)d_in[8];
  const float* wv_b = (const float*)d_in[9];
  const float* wd_w = (const float*)d_in[10];
  const float* wd_b = (const float*)d_in[11];

  const size_t DDsz = (size_t)Dd * Dd;      // 1M
  const size_t BSD = (size_t)Bb * Ss * Dd;  // 8M
  bf16_t* ws = (bf16_t*)d_ws;
  bf16_t* wq_bf = ws;
  bf16_t* wk_bf = wq_bf + DDsz;
  bf16_t* wv_bf = wk_bf + DDsz;
  bf16_t* wd_bf = wv_bf + DDsz;
  bf16_t* Qh = wd_bf + DDsz;
  bf16_t* Kh = Qh + BSD;
  bf16_t* Vh = Kh + BSD;
  bf16_t* Cc = Vh + BSD;

  {
    const int n4 = (int)(DDsz / 4);
    const int blk = 256, grd = (n4 + blk - 1) / blk;
    cvt_f32_bf16_kernel<<<grd, blk, 0, stream>>>(wq_w, wq_bf, n4);
    cvt_f32_bf16_kernel<<<grd, blk, 0, stream>>>(wk_w, wk_bf, n4);
    cvt_f32_bf16_kernel<<<grd, blk, 0, stream>>>(wv_w, wv_bf, n4);
    cvt_f32_bf16_kernel<<<grd, blk, 0, stream>>>(wd_w, wd_bf, n4);
  }

  dim3 ggrid(Dd / 128, (Bb * Ss) / 128);  // (8, 64)
  gemm_xwt_kernel<float, bf16_t><<<ggrid, 256, 0, stream>>>(q, wq_bf, wq_b, Qh, Bb * Ss, Dd, Dd);
  gemm_xwt_kernel<float, bf16_t><<<ggrid, 256, 0, stream>>>(k, wk_bf, wk_b, Kh, Bb * Ss, Dd, Dd);
  gemm_xwt_kernel<float, bf16_t><<<ggrid, 256, 0, stream>>>(v, wv_bf, wv_b, Vh, Bb * Ss, Dd, Dd);

  flash_attn_kernel<<<dim3(Ss / 128, Bb * Hh), 256, 0, stream>>>(Qh, Kh, Vh, Cc);

  gemm_xwt_kernel<bf16_t, float><<<ggrid, 256, 0, stream>>>(Cc, wd_bf, wd_b, (float*)d_out,
                                                            Bb * Ss, Dd, Dd);
}
